// ProteinBindingGNN_3238405342012
// MI455X (gfx1250) — matrix operand order repaired and checked by reading, not compiled, not run on hardware
//
#include <hip/hip_runtime.h>
#include <hip/hip_bf16.h>
#include <math.h>

typedef __attribute__((ext_vector_type(16))) _Float16 v16h;
typedef __attribute__((ext_vector_type(8)))  _Float16 v8h;
typedef __attribute__((ext_vector_type(4)))  _Float16 v4h;
typedef __attribute__((ext_vector_type(8)))  float    v8f;

#define NNODES 2048
#define NEDGES 65536
#define HDIM   256
#define NHEADS 8
#define HEADD  32
#define NIN    1304
#define EIN    4
#define LEG    3
#define LEV    4
#define FFND   1024

enum { ACT_NONE = 0, ACT_RELU = 1, ACT_SILU = 2, ACT_GELU = 3 };

__device__ __forceinline__ float act_apply(float v, int act) {
    if (act == ACT_RELU) return fmaxf(v, 0.f);
    if (act == ACT_SILU) return v * (1.f / (1.f + __expf(-v)));
    if (act == ACT_GELU) return 0.5f * v * (1.f + erff(v * 0.70710678118654752f));
    return v;
}

// ---------------------------------------------------------------------------
// WMMA GEMM: C[M,N] = epilogue(alpha * A[M,K] @ B[K,N])
// Requirements by construction: M % 128 == 0 (all call sites: M = 2048/65536).
// N, K arbitrary (guarded edge-tile paths).
//
// Block: 256 threads = 8 wave32s. Block tile 128(M) x 32(N), K step 32.
// Wave w owns rows [w*16, w*16+16); 2 x v_wmma_f32_16x16x32_f16 per K step,
// sharing one A fragment.
//
// A fragment loaded straight from global as 4x float4 per lane (row pointer /
// gather index hoisted out of the K loop).  B tile staged transposed in LDS
// ([n][k], stride 40 halves) so each B fragment is two 16B ds_load_b128.
// B staging has a branch-free vectorized fast path (full tile in range) and a
// guarded elementwise path only for edge tiles.
// ---------------------------------------------------------------------------
__global__ __launch_bounds__(256) void wmma_gemm(
    const float* __restrict__ A, int lda, const int* __restrict__ aidx,
    const float* __restrict__ B, int ldb, int btrans,
    float* __restrict__ C, int ldc,
    const float* __restrict__ bias,
    const float* __restrict__ resid, int ldres,
    int M, int N, int K, float alpha, int acc, int act)
{
    __shared__ _Float16 sBf[32 * 40];   // B^T tile: [n_local][k_local]

    const int tid    = threadIdx.x;
    const int lane   = tid & 31;
    const int wave   = tid >> 5;
    const int halfid = lane >> 4;       // K-half selector
    const int l16    = lane & 15;
    const int m0     = blockIdx.y * 128;
    const int n0     = blockIdx.x * 32;

    // Staging thread coordinates (fixed per thread).
    const int tI  = tid >> 3;           // 0..31
    const int tJ4 = (tid & 7) * 4;      // 0,4,..,28

    // Per-lane A row pointer; gather index resolved once (M % 128 == 0).
    const int  arowi = m0 + wave * 16 + l16;
    const int  srow  = aidx ? aidx[arowi] : arowi;
    const float* __restrict__ arow = A + (long)srow * lda;

    const bool fullN = (n0 + 32 <= N);

    v8f c0 = {0.f, 0.f, 0.f, 0.f, 0.f, 0.f, 0.f, 0.f};
    v8f c1 = {0.f, 0.f, 0.f, 0.f, 0.f, 0.f, 0.f, 0.f};

    for (int k0 = 0; k0 < K; k0 += 32) {
        // ---- stage B tile (transposed) into LDS ----
        if (fullN && (k0 + 32 <= K)) {
            if (btrans) {
                // B stored [N][K]: float4 along K, one contiguous v4h LDS store
                const float4 bv = *(const float4*)(B + (long)(n0 + tI) * ldb + k0 + tJ4);
                v4h p;
                p[0] = (_Float16)bv.x; p[1] = (_Float16)bv.y;
                p[2] = (_Float16)bv.z; p[3] = (_Float16)bv.w;
                *(v4h*)&sBf[tI * 40 + tJ4] = p;
            } else {
                // B stored [K][N]: float4 along N, 4 b16 LDS stores
                const float4 bv = *(const float4*)(B + (long)(k0 + tI) * ldb + n0 + tJ4);
                sBf[(tJ4 + 0) * 40 + tI] = (_Float16)bv.x;
                sBf[(tJ4 + 1) * 40 + tI] = (_Float16)bv.y;
                sBf[(tJ4 + 2) * 40 + tI] = (_Float16)bv.z;
                sBf[(tJ4 + 3) * 40 + tI] = (_Float16)bv.w;
            }
        } else {
            for (int i = tid; i < 32 * 32; i += 256) {
                int cc = i >> 5, r = i & 31;      // cc = n_local, r = k_local
                int gk = k0 + r, gn = n0 + cc;
                float v = 0.f;
                if (gk < K && gn < N)
                    v = btrans ? B[(long)gn * ldb + gk] : B[(long)gk * ldb + gn];
                sBf[cc * 40 + r] = (_Float16)v;
            }
        }
        __syncthreads();

        // ---- A fragment: 4x float4 straight from global ----
        v16h af;
        if (k0 + 32 <= K) {
            const float4 a0 = *(const float4*)(arow + k0 + halfid * 8);
            const float4 a1 = *(const float4*)(arow + k0 + halfid * 8 + 4);
            const float4 a2 = *(const float4*)(arow + k0 + 16 + halfid * 8);
            const float4 a3 = *(const float4*)(arow + k0 + 16 + halfid * 8 + 4);
            af[0]  = (_Float16)a0.x; af[1]  = (_Float16)a0.y;
            af[2]  = (_Float16)a0.z; af[3]  = (_Float16)a0.w;
            af[4]  = (_Float16)a1.x; af[5]  = (_Float16)a1.y;
            af[6]  = (_Float16)a1.z; af[7]  = (_Float16)a1.w;
            af[8]  = (_Float16)a2.x; af[9]  = (_Float16)a2.y;
            af[10] = (_Float16)a2.z; af[11] = (_Float16)a2.w;
            af[12] = (_Float16)a3.x; af[13] = (_Float16)a3.y;
            af[14] = (_Float16)a3.z; af[15] = (_Float16)a3.w;
            if (k0 + 32 < K) __builtin_prefetch(arow + k0 + 32, 0, 1);  // uniform
        } else {
            #pragma unroll
            for (int e = 0; e < 8; ++e) {
                int k1 = k0 + halfid * 8 + e;
                int k2 = k0 + 16 + halfid * 8 + e;
                af[e]     = (_Float16)((k1 < K) ? arow[k1] : 0.f);
                af[8 + e] = (_Float16)((k2 < K) ? arow[k2] : 0.f);
            }
        }

        // ---- B fragments: two 16B LDS vector loads each ----
        const v8h b0lo = *(const v8h*)&sBf[l16 * 40 + halfid * 8];
        const v8h b0hi = *(const v8h*)&sBf[l16 * 40 + 16 + halfid * 8];
        const v8h b1lo = *(const v8h*)&sBf[(16 + l16) * 40 + halfid * 8];
        const v8h b1hi = *(const v8h*)&sBf[(16 + l16) * 40 + 16 + halfid * 8];
        v16h bf0 = __builtin_shufflevector(b0lo, b0hi, 0,1,2,3,4,5,6,7,8,9,10,11,12,13,14,15);
        v16h bf1 = __builtin_shufflevector(b1lo, b1hi, 0,1,2,3,4,5,6,7,8,9,10,11,12,13,14,15);

        c0 = __builtin_amdgcn_wmma_f32_16x16x32_f16(false, af, false, bf0, (short)0, c0, false, false);
        c1 = __builtin_amdgcn_wmma_f32_16x16x32_f16(false, af, false, bf1, (short)0, c1, false, false);
        __syncthreads();
    }

    // ---- epilogue: f32 C/D layout (VGPR r -> row halfid*8 + r) ----
    #pragma unroll
    for (int g = 0; g < 2; ++g) {
        int gcol = n0 + g * 16 + l16;
        if (gcol < N) {
            #pragma unroll
            for (int r = 0; r < 8; ++r) {
                int grow = m0 + wave * 16 + halfid * 8 + r;
                float v = alpha * (g ? c1[r] : c0[r]);
                long off = (long)grow * ldc + gcol;
                if (acc)   v += C[off];
                if (bias)  v += bias[gcol];
                v = act_apply(v, act);
                if (resid) v += resid[(long)grow * ldres + gcol];
                C[off] = v;
            }
        }
    }
}

// ---------------------------------------------------------------------------
// LayerNorm over 256 features, optional residual input and trailing ReLU.
// ---------------------------------------------------------------------------
__global__ __launch_bounds__(256) void ln256(
    const float* __restrict__ x, const float* __restrict__ res,
    const float* __restrict__ g, const float* __restrict__ be,
    float* __restrict__ out, int relu)
{
    __shared__ float red[256];
    int row = blockIdx.x, j = threadIdx.x;
    long o = (long)row * 256 + j;
    float v = x[o] + (res ? res[o] : 0.f);
    red[j] = v; __syncthreads();
    for (int s2 = 128; s2 > 0; s2 >>= 1) { if (j < s2) red[j] += red[j + s2]; __syncthreads(); }
    float mu = red[0] * (1.f / 256.f);
    __syncthreads();
    float d = v - mu;
    red[j] = d * d; __syncthreads();
    for (int s2 = 128; s2 > 0; s2 >>= 1) { if (j < s2) red[j] += red[j + s2]; __syncthreads(); }
    float var = red[0] * (1.f / 256.f);
    float r = d * rsqrtf(var + 1e-5f) * g[j] + be[j];
    if (relu) r = fmaxf(r, 0.f);
    out[o] = r;
}

// Softmax over one row of length ncols (attention keys).
__global__ __launch_bounds__(256) void softmax_row(float* __restrict__ s, int ncols)
{
    __shared__ float red[256];
    int row = blockIdx.x, tid = threadIdx.x;
    float* p = s + (long)row * ncols;
    float mx = -3.4e38f;
    for (int c = tid; c < ncols; c += 256) mx = fmaxf(mx, p[c]);
    red[tid] = mx; __syncthreads();
    for (int s2 = 128; s2 > 0; s2 >>= 1) { if (tid < s2) red[tid] = fmaxf(red[tid], red[tid + s2]); __syncthreads(); }
    mx = red[0]; __syncthreads();
    float sum = 0.f;
    for (int c = tid; c < ncols; c += 256) { float e = __expf(p[c] - mx); p[c] = e; sum += e; }
    red[tid] = sum; __syncthreads();
    for (int s2 = 128; s2 > 0; s2 >>= 1) { if (tid < s2) red[tid] += red[tid + s2]; __syncthreads(); }
    float inv = 1.f / red[0];
    for (int c = tid; c < ncols; c += 256) p[c] *= inv;
}

__global__ void edge_geom(const float* __restrict__ pos, const int* __restrict__ row,
                          const int* __restrict__ col, float* __restrict__ rel,
                          float* __restrict__ dist, int E)
{
    int e = blockIdx.x * blockDim.x + threadIdx.x;
    if (e >= E) return;
    int rs = row[e], cs = col[e];
    float dx = pos[rs * 3 + 0] - pos[cs * 3 + 0];
    float dy = pos[rs * 3 + 1] - pos[cs * 3 + 1];
    float dz = pos[rs * 3 + 2] - pos[cs * 3 + 2];
    rel[e * 3 + 0] = dx; rel[e * 3 + 1] = dy; rel[e * 3 + 2] = dz;
    dist[e] = fmaxf(sqrtf(dx * dx + dy * dy + dz * dz), 1e-6f);
}

// m1 += dist * mW1[512,:] + bias, then SiLU  (finishes the concat GEMM)
__global__ void m1_finish(float* __restrict__ m1, const float* __restrict__ dist,
                          const float* __restrict__ wd, const float* __restrict__ b, long total)
{
    long t = (long)blockIdx.x * blockDim.x + threadIdx.x;
    if (t >= total) return;
    int j = (int)(t & 255); long e = t >> 8;
    float v = m1[t] + dist[e] * wd[j] + b[j];
    m1[t] = v * (1.f / (1.f + __expf(-v)));
}

__global__ void scatter_add_h(const float* __restrict__ m, const int* __restrict__ col,
                              float* __restrict__ agg, long total)
{
    long t = (long)blockIdx.x * blockDim.x + threadIdx.x;
    if (t >= total) return;
    long e = t >> 8; int j = (int)(t & 255);
    atomicAdd(&agg[(long)col[e] * 256 + j], m[t]);
}

__global__ void scatter_add_c(const float* __restrict__ rel, const float* __restrict__ cw,
                              const int* __restrict__ col, float* __restrict__ agg, int E)
{
    int e = blockIdx.x * blockDim.x + threadIdx.x;
    if (e >= E) return;
    float w = cw[e]; int c = col[e];
    atomicAdd(&agg[c * 3 + 0], rel[e * 3 + 0] * w);
    atomicAdd(&agg[c * 3 + 1], rel[e * 3 + 1] * w);
    atomicAdd(&agg[c * 3 + 2], rel[e * 3 + 2] * w);
}

__global__ void scatter_bias(float* __restrict__ s, const float* __restrict__ bias,
                             const int* __restrict__ row, const int* __restrict__ col,
                             int head, int E, int ncols)
{
    int e = blockIdx.x * blockDim.x + threadIdx.x;
    if (e >= E) return;
    atomicAdd(&s[(long)row[e] * ncols + col[e]], bias[(long)e * NHEADS + head]);
}

__global__ void vec_copy(float* __restrict__ o, const float* __restrict__ i, long n)
{ long t = (long)blockIdx.x * blockDim.x + threadIdx.x; if (t < n) o[t] = i[t]; }
__global__ void vec_add(float* __restrict__ o, const float* __restrict__ i, long n)
{ long t = (long)blockIdx.x * blockDim.x + threadIdx.x; if (t < n) o[t] += i[t]; }
__global__ void vec_zero(float* __restrict__ o, long n)
{ long t = (long)blockIdx.x * blockDim.x + threadIdx.x; if (t < n) o[t] = 0.f; }

// ---------------------------------------------------------------------------

static void launch_gemm(hipStream_t st,
                        const float* A, int lda, const int* aidx,
                        const float* B, int ldb, int btrans,
                        float* C, int ldc, const float* bias,
                        const float* resid, int ldres,
                        int M, int N, int K, float alpha, int acc, int act)
{
    dim3 grid((unsigned)((N + 31) / 32), (unsigned)((M + 127) / 128));
    wmma_gemm<<<grid, 256, 0, st>>>(A, lda, aidx, B, ldb, btrans, C, ldc,
                                    bias, resid, ldres, M, N, K, alpha, acc, act);
}

extern "C" void kernel_launch(void* const* d_in, const int* in_sizes, int n_in,
                              void* d_out, int out_size, void* d_ws, size_t ws_size,
                              hipStream_t stream)
{
    (void)in_sizes; (void)n_in; (void)out_size; (void)ws_size;
    const float* x      = (const float*)d_in[0];
    const float* pos_in = (const float*)d_in[1];
    const int*   eidx   = (const int*)d_in[2];
    const float* eattr  = (const float*)d_in[3];
    const float* np_W   = (const float*)d_in[4];
    const float* np_b   = (const float*)d_in[5];
    const float* np_g   = (const float*)d_in[6];
    const float* np_be  = (const float*)d_in[7];
    const float* ep_W   = (const float*)d_in[8];
    const float* ep_b   = (const float*)d_in[9];
    const float* ep_g   = (const float*)d_in[10];
    const float* ep_be  = (const float*)d_in[11];
    const float* eg_mW1 = (const float*)d_in[12];
    const float* eg_mb1 = (const float*)d_in[13];
    const float* eg_mW2 = (const float*)d_in[14];
    const float* eg_mb2 = (const float*)d_in[15];
    const float* eg_cW1 = (const float*)d_in[16];
    const float* eg_cb1 = (const float*)d_in[17];
    const float* eg_cW2 = (const float*)d_in[18];
    const float* eg_nW1 = (const float*)d_in[19];
    const float* eg_nb1 = (const float*)d_in[20];
    const float* eg_nW2 = (const float*)d_in[21];
    const float* eg_nb2 = (const float*)d_in[22];
    const float* eg_g   = (const float*)d_in[23];
    const float* eg_b   = (const float*)d_in[24];
    const float* ev_qW  = (const float*)d_in[25];
    const float* ev_qb  = (const float*)d_in[26];
    const float* ev_kW  = (const float*)d_in[27];
    const float* ev_kb  = (const float*)d_in[28];
    const float* ev_vW  = (const float*)d_in[29];
    const float* ev_vb  = (const float*)d_in[30];
    const float* ev_oW  = (const float*)d_in[31];
    const float* ev_ob  = (const float*)d_in[32];
    const float* ev_eW  = (const float*)d_in[33];
    const float* ev_eb  = (const float*)d_in[34];
    const float* ev_fW1 = (const float*)d_in[35];
    const float* ev_fb1 = (const float*)d_in[36];
    const float* ev_fW2 = (const float*)d_in[37];
    const float* ev_fb2 = (const float*)d_in[38];
    const float* ev_g1  = (const float*)d_in[39];
    const float* ev_b1  = (const float*)d_in[40];
    const float* ev_g2  = (const float*)d_in[41];
    const float* ev_b2  = (const float*)d_in[42];
    const float* fin_g  = (const float*)d_in[43];
    const float* fin_b  = (const float*)d_in[44];
    const float* hd_W1  = (const float*)d_in[45];
    const float* hd_b1  = (const float*)d_in[46];
    const float* hd_W2  = (const float*)d_in[47];
    const float* hd_b2  = (const float*)d_in[48];

    const int* erow = eidx;            // source j
    const int* ecol = eidx + NEDGES;   // target i

    // ---- workspace carve-up (floats) ----
    float* w = (float*)d_ws;
    float* h    = w; w += (long)NNODES * HDIM;
    float* hn   = w; w += (long)NNODES * HDIM;
    float* t2   = w; w += (long)NNODES * HDIM;   // attn / hn2 scratch
    float* h2   = w; w += (long)NNODES * HDIM;
    float* Qb   = w; w += (long)NNODES * HDIM;
    float* Kb   = w; w += (long)NNODES * HDIM;
    float* Vb   = w; w += (long)NNODES * HDIM;
    float* hagg = w; w += (long)NNODES * HDIM;
    float* cagg = w; w += (long)NNODES * 3;
    float* posb = w; w += (long)NNODES * 3;
    float* ffn1 = w; w += (long)NNODES * FFND;
    float* ht   = w; w += (long)NNODES * (HDIM / 2);
    float* sbuf = w; w += (long)NNODES * NNODES;
    float* ebuf = w; w += (long)NEDGES * HDIM;
    float* m1   = w; w += (long)NEDGES * HDIM;
    float* m2   = w; w += (long)NEDGES * HDIM;
    float* rel  = w; w += (long)NEDGES * 3;
    float* dist = w; w += (long)NEDGES;
    float* bE   = w; w += (long)NEDGES * NHEADS;
    float* cw   = w; w += (long)NEDGES;

    const int TPB = 256;
    #define GRID1(n) dim3((unsigned)(((n) + TPB - 1) / TPB))

    // ---- input projections ----
    launch_gemm(stream, x, NIN, nullptr, np_W, HDIM, 0, h, HDIM, np_b, nullptr, 0,
                NNODES, HDIM, NIN, 1.f, 0, ACT_NONE);
    ln256<<<NNODES, TPB, 0, stream>>>(h, nullptr, np_g, np_be, h, 1);

    launch_gemm(stream, eattr, EIN, nullptr, ep_W, HDIM, 0, ebuf, HDIM, ep_b, nullptr, 0,
                NEDGES, HDIM, EIN, 1.f, 0, ACT_NONE);
    ln256<<<NEDGES, TPB, 0, stream>>>(ebuf, nullptr, ep_g, ep_be, ebuf, 1);

    vec_copy<<<GRID1((long)NNODES * 3), TPB, 0, stream>>>(posb, pos_in, (long)NNODES * 3);

    // ---- EGNN layers ----
    for (int i = 0; i < LEG; ++i) {
        const float* mW1 = eg_mW1 + (long)i * 769 * HDIM;
        const float* mW2 = eg_mW2 + (long)i * HDIM * HDIM;
        const float* cW1 = eg_cW1 + (long)i * HDIM * HDIM;
        const float* cW2 = eg_cW2 + (long)i * HDIM * 1;
        const float* nW1 = eg_nW1 + (long)i * 512 * HDIM;
        const float* nW2 = eg_nW2 + (long)i * HDIM * HDIM;

        edge_geom<<<GRID1(NEDGES), TPB, 0, stream>>>(posb, erow, ecol, rel, dist, NEDGES);

        // m1 = [h_i, h_j, dist, e] @ mW1 as 3 gather passes + rank-1 finish
        launch_gemm(stream, h, HDIM, ecol, mW1,              HDIM, 0, m1, HDIM, nullptr, nullptr, 0,
                    NEDGES, HDIM, HDIM, 1.f, 0, ACT_NONE);
        launch_gemm(stream, h, HDIM, erow, mW1 + 256 * HDIM, HDIM, 0, m1, HDIM, nullptr, nullptr, 0,
                    NEDGES, HDIM, HDIM, 1.f, 1, ACT_NONE);
        launch_gemm(stream, ebuf, HDIM, nullptr, mW1 + 513 * HDIM, HDIM, 0, m1, HDIM, nullptr, nullptr, 0,
                    NEDGES, HDIM, HDIM, 1.f, 1, ACT_NONE);
        m1_finish<<<GRID1((long)NEDGES * HDIM), TPB, 0, stream>>>(
            m1, dist, mW1 + 512 * HDIM, eg_mb1 + (long)i * HDIM, (long)NEDGES * HDIM);

        launch_gemm(stream, m1, HDIM, nullptr, mW2, HDIM, 0, m2, HDIM, eg_mb2 + (long)i * HDIM, nullptr, 0,
                    NEDGES, HDIM, HDIM, 1.f, 0, ACT_SILU);
        launch_gemm(stream, m2, HDIM, nullptr, cW1, HDIM, 0, m1, HDIM, eg_cb1 + (long)i * HDIM, nullptr, 0,
                    NEDGES, HDIM, HDIM, 1.f, 0, ACT_SILU);
        launch_gemm(stream, m1, HDIM, nullptr, cW2, 1, 0, cw, 1, nullptr, nullptr, 0,
                    NEDGES, 1, HDIM, 1.f, 0, ACT_NONE);

        vec_zero<<<GRID1((long)NNODES * HDIM), TPB, 0, stream>>>(hagg, (long)NNODES * HDIM);
        vec_zero<<<GRID1((long)NNODES * 3), TPB, 0, stream>>>(cagg, (long)NNODES * 3);
        scatter_add_h<<<GRID1((long)NEDGES * HDIM), TPB, 0, stream>>>(m2, ecol, hagg, (long)NEDGES * HDIM);
        scatter_add_c<<<GRID1(NEDGES), TPB, 0, stream>>>(rel, cw, ecol, cagg, NEDGES);

        // node MLP: silu([h, h_agg] @ nW1 + nb1) @ nW2 + nb2
        launch_gemm(stream, h, HDIM, nullptr, nW1, HDIM, 0, Qb, HDIM, nullptr, nullptr, 0,
                    NNODES, HDIM, HDIM, 1.f, 0, ACT_NONE);
        launch_gemm(stream, hagg, HDIM, nullptr, nW1 + 256 * HDIM, HDIM, 0, Qb, HDIM,
                    eg_nb1 + (long)i * HDIM, nullptr, 0, NNODES, HDIM, HDIM, 1.f, 1, ACT_SILU);
        launch_gemm(stream, Qb, HDIM, nullptr, nW2, HDIM, 0, t2, HDIM, eg_nb2 + (long)i * HDIM, nullptr, 0,
                    NNODES, HDIM, HDIM, 1.f, 0, ACT_NONE);

        ln256<<<NNODES, TPB, 0, stream>>>(h, t2, eg_g + (long)i * HDIM, eg_b + (long)i * HDIM, h, 0);
        vec_add<<<GRID1((long)NNODES * 3), TPB, 0, stream>>>(posb, cagg, (long)NNODES * 3);
    }

    // ---- Evoformer layers ----
    float* hcur = h;
    float* halt = h2;
    const float scale = 0.17677669529663687f;  // HD^-0.5
    for (int i = 0; i < LEV; ++i) {
        ln256<<<NNODES, TPB, 0, stream>>>(hcur, nullptr, ev_g1 + (long)i * HDIM, ev_b1 + (long)i * HDIM, hn, 0);
        launch_gemm(stream, hn, HDIM, nullptr, ev_qW + (long)i * HDIM * HDIM, HDIM, 0, Qb, HDIM,
                    ev_qb + (long)i * HDIM, nullptr, 0, NNODES, HDIM, HDIM, 1.f, 0, ACT_NONE);
        launch_gemm(stream, hn, HDIM, nullptr, ev_kW + (long)i * HDIM * HDIM, HDIM, 0, Kb, HDIM,
                    ev_kb + (long)i * HDIM, nullptr, 0, NNODES, HDIM, HDIM, 1.f, 0, ACT_NONE);
        launch_gemm(stream, hn, HDIM, nullptr, ev_vW + (long)i * HDIM * HDIM, HDIM, 0, Vb, HDIM,
                    ev_vb + (long)i * HDIM, nullptr, 0, NNODES, HDIM, HDIM, 1.f, 0, ACT_NONE);
        launch_gemm(stream, ebuf, HDIM, nullptr, ev_eW + (long)i * HDIM * NHEADS, NHEADS, 0, bE, NHEADS,
                    ev_eb + (long)i * NHEADS, nullptr, 0, NEDGES, NHEADS, HDIM, 1.f, 0, ACT_NONE);

        for (int hd = 0; hd < NHEADS; ++hd) {
            // S = scale * Q_h @ K_h^T  (B transposed)
            launch_gemm(stream, Qb + hd * HEADD, HDIM, nullptr, Kb + hd * HEADD, HDIM, 1,
                        sbuf, NNODES, nullptr, nullptr, 0,
                        NNODES, NNODES, HEADD, scale, 0, ACT_NONE);
            scatter_bias<<<GRID1(NEDGES), TPB, 0, stream>>>(sbuf, bE, erow, ecol, hd, NEDGES, NNODES);
            softmax_row<<<NNODES, TPB, 0, stream>>>(sbuf, NNODES);
            // attn_h = S @ V_h
            launch_gemm(stream, sbuf, NNODES, nullptr, Vb + hd * HEADD, HDIM, 0,
                        t2 + hd * HEADD, HDIM, nullptr, nullptr, 0,
                        NNODES, HEADD, NNODES, 1.f, 0, ACT_NONE);
        }

        // h = h + attn @ oW + ob
        launch_gemm(stream, t2, HDIM, nullptr, ev_oW + (long)i * HDIM * HDIM, HDIM, 0, halt, HDIM,
                    ev_ob + (long)i * HDIM, hcur, HDIM, NNODES, HDIM, HDIM, 1.f, 0, ACT_NONE);
        { float* t = hcur; hcur = halt; halt = t; }

        // FFN with pre-LN, exact GELU, residual
        ln256<<<NNODES, TPB, 0, stream>>>(hcur, nullptr, ev_g2 + (long)i * HDIM, ev_b2 + (long)i * HDIM, hn, 0);
        launch_gemm(stream, hn, HDIM, nullptr, ev_fW1 + (long)i * HDIM * FFND, FFND, 0, ffn1, FFND,
                    ev_fb1 + (long)i * FFND, nullptr, 0, NNODES, FFND, HDIM, 1.f, 0, ACT_GELU);
        launch_gemm(stream, ffn1, FFND, nullptr, ev_fW2 + (long)i * FFND * HDIM, HDIM, 0, halt, HDIM,
                    ev_fb2 + (long)i * HDIM, hcur, HDIM, NNODES, HDIM, FFND, 1.f, 0, ACT_NONE);
        { float* t = hcur; hcur = halt; halt = t; }
    }

    // ---- final LN + head ----
    ln256<<<NNODES, TPB, 0, stream>>>(hcur, nullptr, fin_g, fin_b, hn, 0);
    launch_gemm(stream, hn, HDIM, nullptr, hd_W1, HDIM / 2, 0, ht, HDIM / 2, hd_b1, nullptr, 0,
                NNODES, HDIM / 2, HDIM, 1.f, 0, ACT_GELU);
    launch_gemm(stream, ht, HDIM / 2, nullptr, hd_W2, 1, 0, (float*)d_out, 1, hd_b2, nullptr, 0,
                NNODES, 1, HDIM / 2, 1.f, 0, ACT_NONE);
    #undef GRID1
}
